// NodeModel_52252572123265
// MI455X (gfx1250) — matrix operand order repaired and checked by reading, not compiled, not run on hardware
//
#include <hip/hip_runtime.h>
#include <hip/hip_bf16.h>

typedef __attribute__((ext_vector_type(16))) _Float16 v16h;
typedef __attribute__((ext_vector_type(8)))  _Float16 v8h;
typedef __attribute__((ext_vector_type(8)))  float    v8f;
typedef __attribute__((ext_vector_type(4)))  float    v4f;

#define N_NODES 50000
#define N_EDGES 800000
#define D_NODE  128
#define D_EDGE  64
#define D_IN    192   // D_NODE + D_EDGE
#define H1      256
#define H2      128
#define ROWS    32    // node rows per block (2 M-tiles)

// ---------------- workspace layout ----------------
#define AGG_BYTES   ((size_t)N_NODES * D_EDGE * 4)        // 12,800,000
#define W1T_OFF     AGG_BYTES
#define W1T_ELEMS   (H1 * D_IN)                            // 49152
#define W2T_OFF     (W1T_OFF + (size_t)W1T_ELEMS * 2)
#define W2T_ELEMS   (H2 * H1)                              // 32768

__device__ __forceinline__ void atomic_add_f32(float* p, float v) {
    unsafeAtomicAdd(p, v);   // native global_atomic_add_f32 on gfx1250
}

// ---------------- kernel 1: zero agg ----------------
__global__ void zero_agg_kernel(float* agg) {
    int tid = blockIdx.x * blockDim.x + threadIdx.x;
    if (tid < N_NODES * D_EDGE) agg[tid] = 0.0f;
}

// ---------------- kernel 2: transpose+convert weights to f16 ----------------
__global__ void convert_weights_kernel(const float* __restrict__ W1,
                                       const float* __restrict__ W2,
                                       _Float16* __restrict__ w1t,
                                       _Float16* __restrict__ w2t) {
    int tid = blockIdx.x * blockDim.x + threadIdx.x;
    if (tid < W1T_ELEMS) {
        int n = tid / D_IN, k = tid % D_IN;       // w1t[n][k] = W1[k][n]
        w1t[tid] = (_Float16)W1[k * H1 + n];
    } else if (tid < W1T_ELEMS + W2T_ELEMS) {
        int t = tid - W1T_ELEMS;
        int n = t / H1, k = t % H1;               // w2t[n][k] = W2[k][n]
        w2t[t] = (_Float16)W2[k * H2 + n];
    }
}

// ---------------- kernel 3: scatter-add edge features ----------------
// one thread per (edge, 4-float chunk): N_EDGES * 16 threads
__global__ void scatter_kernel(const float* __restrict__ edge_attr,
                               const int* __restrict__ edge_index,
                               float* __restrict__ agg) {
    int tid = blockIdx.x * blockDim.x + threadIdx.x;
    if (tid >= N_EDGES * (D_EDGE / 4)) return;
    int e  = tid >> 4;          // D_EDGE/4 == 16 chunks per edge
    int c4 = tid & 15;
    int dst = edge_index[N_EDGES + e];            // col = edge_index[1][e]
    // edge_attr is read exactly once: non-temporal so it doesn't evict
    // the L2-resident agg buffer.
    v4f v = __builtin_nontemporal_load((const v4f*)edge_attr + tid);
    float* base = agg + (size_t)dst * D_EDGE + c4 * 4;
    atomic_add_f32(base + 0, v.x);
    atomic_add_f32(base + 1, v.y);
    atomic_add_f32(base + 2, v.z);
    atomic_add_f32(base + 3, v.w);
}

// ---------------- kernel 4: fused 2-layer MLP via WMMA ----------------
// block = 128 threads (4 waves), 32 node rows (two 16-row M-tiles) per block.
// Each B fragment feeds 2 WMMAs (one per M-subtile).
__global__ __launch_bounds__(128)
void mlp_kernel(const float* __restrict__ x,
                const float* __restrict__ agg,
                const _Float16* __restrict__ w1t,
                const float* __restrict__ b1,
                const _Float16* __restrict__ w2t,
                const float* __restrict__ b2,
                float* __restrict__ out) {
    __shared__ __align__(16) _Float16 Atile[ROWS * D_IN];   // 12 KB
    __shared__ __align__(16) _Float16 Htile[ROWS * H1];     // 16 KB

    const int wave = threadIdx.x >> 5;
    const int lane = threadIdx.x & 31;
    const int m0   = blockIdx.x * ROWS;

    // ---- stage A tile: concat(x, agg) rows m0..m0+31 as f16 (pad OOB with 0) ----
    for (int idx = threadIdx.x; idx < ROWS * D_IN; idx += 128) {
        int r = idx / D_IN, c = idx % D_IN;
        int m = m0 + r;
        float v = 0.0f;
        if (m < N_NODES)
            v = (c < D_NODE) ? x[(size_t)m * D_NODE + c]
                             : agg[(size_t)m * D_EDGE + (c - D_NODE)];
        Atile[idx] = (_Float16)v;
    }
    __syncthreads();

    const int arow = lane & 15;        // M index within a 16-row subtile
    const int ahi  = lane >> 4;        // 0/1: which 16-wide K half (contiguous k order)

    // ---- GEMM1: [32 x 192] x [192 x 256], each wave owns 4 N-tiles x 2 M-tiles ----
    v8f acc[2][4] = {{v8f{}, v8f{}, v8f{}, v8f{}}, {v8f{}, v8f{}, v8f{}, v8f{}}};
    for (int ks = 0; ks < D_IN; ks += 32) {
        v16h af[2];
        #pragma unroll
        for (int mt = 0; mt < 2; mt++) {
            int row = mt * 16 + arow;
            v8h a_lo = *(const v8h*)&Atile[row * D_IN + ks + ahi * 16];
            v8h a_hi = *(const v8h*)&Atile[row * D_IN + ks + ahi * 16 + 8];
            #pragma unroll
            for (int i = 0; i < 8; i++) { af[mt][i] = a_lo[i]; af[mt][8 + i] = a_hi[i]; }
        }
        #pragma unroll
        for (int t = 0; t < 4; t++) {
            int bcol = (wave * 4 + t) * 16 + (lane & 15);
            int kb   = ks + (lane >> 4) * 16;
            v16h bfrag = *(const v16h*)&w1t[bcol * D_IN + kb];  // contiguous 32B
            acc[0][t] = __builtin_amdgcn_wmma_f32_16x16x32_f16(
                false, af[0], false, bfrag, (short)0, acc[0][t], false, false);
            acc[1][t] = __builtin_amdgcn_wmma_f32_16x16x32_f16(
                false, af[1], false, bfrag, (short)0, acc[1][t], false, false);
        }
    }

    // ---- bias + relu -> Htile (f16, row-major 32 x 256) ----
    #pragma unroll
    for (int t = 0; t < 4; t++) {
        int ncol = (wave * 4 + t) * 16 + (lane & 15);
        float bias = b1[ncol];
        #pragma unroll
        for (int mt = 0; mt < 2; mt++) {
            #pragma unroll
            for (int r = 0; r < 8; r++) {
                int m = mt * 16 + r + (lane >> 4) * 8;
                float v = acc[mt][t][r] + bias;
                v = v > 0.0f ? v : 0.0f;
                Htile[m * H1 + ncol] = (_Float16)v;
            }
        }
    }
    __syncthreads();

    // ---- GEMM2: [32 x 256] x [256 x 128], each wave owns 2 N-tiles x 2 M-tiles ----
    v8f acc2[2][2] = {{v8f{}, v8f{}}, {v8f{}, v8f{}}};
    for (int ks = 0; ks < H1; ks += 32) {
        v16h af[2];
        #pragma unroll
        for (int mt = 0; mt < 2; mt++) {
            int row = mt * 16 + arow;
            v8h a_lo = *(const v8h*)&Htile[row * H1 + ks + ahi * 16];
            v8h a_hi = *(const v8h*)&Htile[row * H1 + ks + ahi * 16 + 8];
            #pragma unroll
            for (int i = 0; i < 8; i++) { af[mt][i] = a_lo[i]; af[mt][8 + i] = a_hi[i]; }
        }
        #pragma unroll
        for (int t = 0; t < 2; t++) {
            int bcol = (wave * 2 + t) * 16 + (lane & 15);
            int kb   = ks + (lane >> 4) * 16;
            v16h bfrag = *(const v16h*)&w2t[bcol * H1 + kb];
            acc2[0][t] = __builtin_amdgcn_wmma_f32_16x16x32_f16(
                false, af[0], false, bfrag, (short)0, acc2[0][t], false, false);
            acc2[1][t] = __builtin_amdgcn_wmma_f32_16x16x32_f16(
                false, af[1], false, bfrag, (short)0, acc2[1][t], false, false);
        }
    }

    // ---- bias + relu -> out (fp32, non-temporal: written once, never reread) ----
    #pragma unroll
    for (int t = 0; t < 2; t++) {
        int ncol = (wave * 2 + t) * 16 + (lane & 15);
        float bias = b2[ncol];
        #pragma unroll
        for (int mt = 0; mt < 2; mt++) {
            #pragma unroll
            for (int r = 0; r < 8; r++) {
                int m = m0 + mt * 16 + r + (lane >> 4) * 8;
                if (m < N_NODES) {
                    float v = acc2[mt][t][r] + bias;
                    v = v > 0.0f ? v : 0.0f;
                    __builtin_nontemporal_store(v, &out[(size_t)m * H2 + ncol]);
                }
            }
        }
    }
}

extern "C" void kernel_launch(void* const* d_in, const int* in_sizes, int n_in,
                              void* d_out, int out_size, void* d_ws, size_t ws_size,
                              hipStream_t stream) {
    const float* x          = (const float*)d_in[0];
    const int*   edge_index = (const int*)  d_in[1];
    const float* edge_attr  = (const float*)d_in[2];
    const float* W1         = (const float*)d_in[3];
    const float* b1         = (const float*)d_in[4];
    const float* W2         = (const float*)d_in[5];
    const float* b2         = (const float*)d_in[6];
    float* out = (float*)d_out;

    char* ws = (char*)d_ws;
    float*    agg = (float*)ws;
    _Float16* w1t = (_Float16*)(ws + W1T_OFF);
    _Float16* w2t = (_Float16*)(ws + W2T_OFF);

    // 1. zero the aggregation buffer (workspace is poisoned)
    {
        int n = N_NODES * D_EDGE;
        zero_agg_kernel<<<(n + 255) / 256, 256, 0, stream>>>(agg);
    }
    // 2. transpose+convert weights to f16 (n-major, k-contiguous)
    {
        int n = W1T_ELEMS + W2T_ELEMS;
        convert_weights_kernel<<<(n + 255) / 256, 256, 0, stream>>>(W1, W2, w1t, w2t);
    }
    // 3. scatter-add edge features onto destination nodes
    {
        int n = N_EDGES * (D_EDGE / 4);
        scatter_kernel<<<(n + 255) / 256, 256, 0, stream>>>(edge_attr, edge_index, agg);
    }
    // 4. fused MLP: 32 rows per block
    mlp_kernel<<<(N_NODES + ROWS - 1) / ROWS, 128, 0, stream>>>(
        x, agg, w1t, b1, w2t, b2, out);
}
